// graph_constructor_11931419148612
// MI455X (gfx1250) — matrix operand order repaired and checked by reading, not compiled, not run on hardware
//
#include <hip/hip_runtime.h>
#include <math.h>

typedef __attribute__((ext_vector_type(16))) _Float16 v16h;
typedef __attribute__((ext_vector_type(8)))  _Float16 v8h;
typedef __attribute__((ext_vector_type(8)))  float    v8f;

#define NNODES 8192
#define DIM    128
#define TOPK   32
#define ALPHA  3.0f

// Branch-free relu(tanh(y)): t = (1-p)/(1+p), p = exp(-2y) = 2^(-y*2*log2(e)).
// Clamp keeps p finite (no Inf/Inf NaN); negative side is removed by the fmax.
// Lowers to v_exp_f32 + v_rcp_f32 + a few VALU ops, no divergent slow path.
__device__ __forceinline__ float relu_tanh(float y) {
  y = fminf(fmaxf(y, -30.0f), 30.0f);
  const float p = __builtin_amdgcn_exp2f(-2.885390081777927f * y);
  const float t = (1.0f - p) * __builtin_amdgcn_rcpf(1.0f + p);
  return fmaxf(t, 0.0f);
}

// ---------------------------------------------------------------------------
// Kernel A: nodevec = tanh(ALPHA * (emb @ lin^T + bias)), emitted as f16.
// One block per node row (128 threads, one per output dim). fp32 math for
// precision before the tanh; lin table (64KB) streams from L2. Cost trivial,
// so keep full-precision tanhf here.
// ---------------------------------------------------------------------------
__global__ void nodevec_kernel(const float* __restrict__ emb,
                               const float* __restrict__ lin,
                               const float* __restrict__ bias,
                               _Float16* __restrict__ outh) {
  __shared__ float s_emb[DIM];
  const int n = blockIdx.x;
  const int d = threadIdx.x;
  s_emb[d] = emb[(size_t)n * DIM + d];
  __syncthreads();
  const float* lrow = lin + (size_t)d * DIM;
  float s = bias[d];
#pragma unroll 8
  for (int k = 0; k < DIM; k += 4) {
    float4 l4 = *(const float4*)(lrow + k);
    s += s_emb[k]     * l4.x;
    s += s_emb[k + 1] * l4.y;
    s += s_emb[k + 2] * l4.z;
    s += s_emb[k + 3] * l4.w;
  }
  outh[(size_t)n * DIM + d] = (_Float16)tanhf(ALPHA * s);
}

// ---------------------------------------------------------------------------
// Kernel B: adj = relu(tanh(ALPHA * (V1 V2^T - V2 V1^T))) via WMMA f16.
// Block = 256 threads = 8 waves; block computes a 128x128 tile, wave w owns a
// 16x128 row stripe (8 16x16 output tiles). K=128 -> 4 chained
// v_wmma_f32_16x16x32_f16 per accumulator; two accumulators per tile (the two
// products), subtracted in VALU. Fragment loads use contiguous k = 16*hs + s
// in both A and B and read straight from the row-major f16 nodevecs (4MB total,
// L2-resident). Store-bandwidth bound: 256MB adj write ~= 11us at 23.3 TB/s.
// ---------------------------------------------------------------------------
__global__ void adj_kernel(const _Float16* __restrict__ V1,
                           const _Float16* __restrict__ V2,
                           float* __restrict__ out) {
  const int lane = threadIdx.x & 31;
  const int wave = threadIdx.x >> 5;
  const int hs   = lane >> 4;      // half-wave select
  const int lm   = lane & 15;
  const int I     = blockIdx.y * 128 + wave * 16;  // row-tile base
  const int Jbase = blockIdx.x * 128;              // col-tile base

  // A fragments: row M = I + lm, two 8-half chunks per k-step.
  const _Float16* pA1 = V1 + (size_t)(I + lm) * DIM;
  const _Float16* pA2 = V2 + (size_t)(I + lm) * DIM;
  v16h a1[4], a2[4];
#pragma unroll
  for (int kk = 0; kk < 4; ++kk) {
    const int k0 = kk * 32 + hs * 16;
    v8h c0 = *(const v8h*)(pA1 + k0);
    v8h c1 = *(const v8h*)(pA1 + k0 + 8);
    v8h d0 = *(const v8h*)(pA2 + k0);
    v8h d1 = *(const v8h*)(pA2 + k0 + 8);
#pragma unroll
    for (int i = 0; i < 8; ++i) {
      a1[kk][i] = c0[i]; a1[kk][8 + i] = c1[i];
      a2[kk][i] = d0[i]; a2[kk][8 + i] = d1[i];
    }
  }

#pragma unroll
  for (int nt = 0; nt < 8; ++nt) {
    const int Jn = Jbase + nt * 16;
    // B fragment: column N = Jn + lm -> row (Jn+lm) of the other nodevec,
    // 16 contiguous K values starting at kk*32 + hs*16.
    const _Float16* pB1 = V2 + (size_t)(Jn + lm) * DIM + hs * 16;
    const _Float16* pB2 = V1 + (size_t)(Jn + lm) * DIM + hs * 16;
    v8f acc1 = {};
    v8f acc2 = {};
#pragma unroll
    for (int kk = 0; kk < 4; ++kk) {
      v16h b1 = *(const v16h*)(pB1 + kk * 32);
      v16h b2 = *(const v16h*)(pB2 + kk * 32);
      acc1 = __builtin_amdgcn_wmma_f32_16x16x32_f16(
          false, a1[kk], false, b1, (short)0, acc1, false, false);
      acc2 = __builtin_amdgcn_wmma_f32_16x16x32_f16(
          false, a2[kk], false, b2, (short)0, acc2, false, false);
    }
#pragma unroll
    for (int r = 0; r < 8; ++r) {
      const int row = I + hs * 8 + r;
      out[(size_t)row * NNODES + Jn + lm] =
          relu_tanh(ALPHA * (acc1[r] - acc2[r]));
    }
  }
}

// ---------------------------------------------------------------------------
// Kernel C: per-row top-32 on (adj + noise*0.01), zero everything else,
// in place on d_out. One block (256 threads) per row; 32 jittered values per
// thread in registers (fully unrolled indexing -> stays in VGPRs; selected
// elements tracked via a bitmask, excluded from rescans).
//
// Lazy argmax: each lane keeps a persistent local (max,idx); per pass only
// the WINNER'S wave re-scans (one lane, predicated) and re-reduces; the other
// waves' LDS entries stay valid. ~6x less VALU than rescanning every pass.
// Tie-break: lower index wins (matches jax top_k ordering).
// ---------------------------------------------------------------------------
__global__ void topk_mask_kernel(float* __restrict__ adj,
                                 const float* __restrict__ noise) {
  const int row = blockIdx.x;
  const int tid = threadIdx.x;                 // 256 threads
  constexpr int VPT = NNODES / 256;            // 32 values per thread
  const size_t base = (size_t)row * NNODES;

  float v[VPT], jit[VPT];
  unsigned sel = 0u;
#pragma unroll
  for (int j = 0; j < VPT; ++j) {
    const int i = tid + j * 256;               // coalesced
    v[j]   = adj[base + i];
    jit[j] = fmaf(noise[base + i], 0.01f, v[j]);
  }

  __shared__ float s_val[8];
  __shared__ int   s_idx[8];
  __shared__ int   s_win;
  const int lane = tid & 31;
  const int wv   = tid >> 5;

  // Persistent per-lane local argmax.
  float bv = -__builtin_inff();
  int   bi = 0x7fffffff;
#pragma unroll
  for (int j = 0; j < VPT; ++j) {
    const int i = tid + j * 256;
    if (jit[j] > bv || (jit[j] == bv && i < bi)) { bv = jit[j]; bi = i; }
  }
  // Initial wave reduction into LDS (keep bv/bi intact; reduce in rv/ri).
  {
    float rv = bv; int ri = bi;
#pragma unroll
    for (int off = 16; off > 0; off >>= 1) {
      const float ov = __shfl_xor(rv, off, 32);
      const int   oi = __shfl_xor(ri, off, 32);
      if (ov > rv || (ov == rv && oi < ri)) { rv = ov; ri = oi; }
    }
    if (lane == 0) { s_val[wv] = rv; s_idx[wv] = ri; }
  }
  __syncthreads();

  for (int pass = 0; pass < TOPK; ++pass) {
    if (tid == 0) {
      float mv = s_val[0]; int mi = s_idx[0];
#pragma unroll
      for (int w = 1; w < 8; ++w) {
        if (s_val[w] > mv || (s_val[w] == mv && s_idx[w] < mi)) {
          mv = s_val[w]; mi = s_idx[w];
        }
      }
      s_win = mi;
    }
    __syncthreads();
    const int wi       = s_win;
    const int ownerTid = wi & 255;
    if ((ownerTid >> 5) == wv) {             // uniform per wave -> cheap skip
      if (tid == ownerTid) {
        sel |= 1u << (wi >> 8);              // runtime shift, no array index
        // Recompute this lane's local max excluding selected elements.
        bv = -__builtin_inff();
        bi = 0x7fffffff;
#pragma unroll
        for (int j = 0; j < VPT; ++j) {
          if (!((sel >> j) & 1u)) {
            const int i = tid + j * 256;
            if (jit[j] > bv || (jit[j] == bv && i < bi)) { bv = jit[j]; bi = i; }
          }
        }
      }
      // Re-reduce this wave from persistent per-lane (bv,bi).
      float rv = bv; int ri = bi;
#pragma unroll
      for (int off = 16; off > 0; off >>= 1) {
        const float ov = __shfl_xor(rv, off, 32);
        const int   oi = __shfl_xor(ri, off, 32);
        if (ov > rv || (ov == rv && oi < ri)) { rv = ov; ri = oi; }
      }
      if (lane == 0) { s_val[wv] = rv; s_idx[wv] = ri; }
    }
    __syncthreads();
  }

#pragma unroll
  for (int j = 0; j < VPT; ++j) {
    const int i = tid + j * 256;
    adj[base + i] = ((sel >> j) & 1u) ? v[j] : 0.0f;
  }
}

// ---------------------------------------------------------------------------
// Launch: inputs are (x, scale_set, emb1_w, emb2_w, lin1_w, lin1_b, lin2_w,
// lin2_b, noise); x/scale_set unused by the reference math. Workspace use:
// 4 MB for the two f16 nodevec tables.
// ---------------------------------------------------------------------------
extern "C" void kernel_launch(void* const* d_in, const int* in_sizes, int n_in,
                              void* d_out, int out_size, void* d_ws, size_t ws_size,
                              hipStream_t stream) {
  const float* emb1  = (const float*)d_in[2];
  const float* emb2  = (const float*)d_in[3];
  const float* lin1  = (const float*)d_in[4];
  const float* b1    = (const float*)d_in[5];
  const float* lin2  = (const float*)d_in[6];
  const float* b2    = (const float*)d_in[7];
  const float* noise = (const float*)d_in[8];
  float* out = (float*)d_out;

  _Float16* V1 = (_Float16*)d_ws;
  _Float16* V2 = V1 + (size_t)NNODES * DIM;

  nodevec_kernel<<<NNODES, DIM, 0, stream>>>(emb1, lin1, b1, V1);
  nodevec_kernel<<<NNODES, DIM, 0, stream>>>(emb2, lin2, b2, V2);
  adj_kernel<<<dim3(NNODES / 128, NNODES / 128), 256, 0, stream>>>(V1, V2, out);
  topk_mask_kernel<<<NNODES, 256, 0, stream>>>(out, noise);
}
